// HyperLSTMCell_78262894067861
// MI455X (gfx1250) — matrix operand order repaired and checked by reading, not compiled, not run on hardware
//
#include <hip/hip_runtime.h>
#include <hip/hip_bf16.h>
#include <math.h>

// ---------------------------------------------------------------------------
// CDNA5 (gfx1250, wave32) types
// ---------------------------------------------------------------------------
typedef __attribute__((ext_vector_type(16))) __bf16 v16bf;
typedef __attribute__((ext_vector_type(8)))  float  v8f;
typedef __attribute__((ext_vector_type(4)))  float  f32x4;
typedef __attribute__((ext_vector_type(4)))  __bf16 bf16x4;
typedef __attribute__((ext_vector_type(4)))  int    v4i;

// Async global->LDS path (CDNA5): guarded so the file always compiles.
#if __has_builtin(__builtin_amdgcn_global_load_async_to_lds_b128) && \
    __has_builtin(__builtin_amdgcn_s_wait_asynccnt)
#define USE_ASYNC_LDS 1
#else
#define USE_ASYNC_LDS 0
#endif

#define BM 128
#define BN 128
#define BK 32
#define BKP 48   // LDS K-stride: 96 B/row keeps every 16B chunk b128-aligned

union FragBF { uint4 q[2]; v16bf v; };
union BFPack { bf16x4 h[2]; uint4 q; };

// ---------------------------------------------------------------------------
// Packed fp32 -> bf16 conversion (one pass per tensor; hardware v_cvt path)
// ---------------------------------------------------------------------------
__global__ __launch_bounds__(256)
void cvt_f32_bf16_kernel(const float* __restrict__ s, __bf16* __restrict__ d,
                         long long n8)
{
  long long i = (long long)blockIdx.x * 256 + threadIdx.x;
  if (i >= n8) return;
  f32x4 a = ((const f32x4*)s)[2 * i];
  f32x4 b = ((const f32x4*)s)[2 * i + 1];
  BFPack p;
  p.h[0] = __builtin_convertvector(a, bf16x4);
  p.h[1] = __builtin_convertvector(b, bf16x4);
  ((uint4*)d)[i] = p.q;
}

// ---------------------------------------------------------------------------
// GEMM:  C[M,N] (+)= A_bf16[M,K] * W_bf16[N,K]^T   (WMMA bf16, f32 acc)
// grid = (N/BN, M/BM, batch); block = 256 threads (8 waves).
// Waves: 2 (M, 64 rows) x 4 (N, 32 cols); per wave 4x2 WMMA 16x16 tiles.
// Double-buffered LDS staging via GLOBAL_LOAD_ASYNC_TO_LDS_B128:
// tile i+1 DMA overlaps tile i WMMA; s_wait_asynccnt(4) drains only tile i.
// ---------------------------------------------------------------------------
__global__ __launch_bounds__(256)
void gemm_bf16_wmma(const __bf16* __restrict__ A, int lda, long long sAb,
                    const __bf16* __restrict__ W, int ldw, long long sWb,
                    float* __restrict__ C, int ldc, long long sCb,
                    int K, int accumulate)
{
  __shared__ __align__(16) __bf16 Alds[2][BM][BKP];
  __shared__ __align__(16) __bf16 Wlds[2][BN][BKP];

  const int tid   = threadIdx.x;
  const int lane  = tid & 31;
  const int wv    = tid >> 5;
  const int waveM = wv >> 2;        // 0..1 -> 64-row slice of BM
  const int waveN = wv & 3;         // 0..3 -> 32-col slice of BN
  const int half  = lane >> 4;      // 0: lanes 0-15, 1: lanes 16-31
  const int l16   = lane & 15;

  const __bf16* Ab = A + (long long)blockIdx.z * sAb + (long long)blockIdx.y * BM * lda;
  const __bf16* Wb = W + (long long)blockIdx.z * sWb + (long long)blockIdx.x * BN * ldw;
  float*        Cb = C + (long long)blockIdx.z * sCb;

  // Stage one BK-slab into LDS buffer `buf` (4 async instructions per wave).
  auto stage = [&](int kk, int buf) {
    #pragma unroll
    for (int i = 0; i < 2; ++i) {
      int idx = tid + i * 256;                 // 0..511
      int r = idx >> 2, c = (idx & 3) * 8;     // 8 bf16 per 16B chunk
      const __bf16* ga = Ab + (long long)r * lda + kk + c;
      const __bf16* gw = Wb + (long long)r * ldw + kk + c;
#if USE_ASYNC_LDS
      __builtin_amdgcn_global_load_async_to_lds_b128(
          (v4i*)ga, (v4i*)&Alds[buf][r][c], 0, 0);
      __builtin_amdgcn_global_load_async_to_lds_b128(
          (v4i*)gw, (v4i*)&Wlds[buf][r][c], 0, 0);
#else
      *(uint4*)&Alds[buf][r][c] = *(const uint4*)ga;
      *(uint4*)&Wlds[buf][r][c] = *(const uint4*)gw;
#endif
    }
  };

  v8f acc[4][2];
  #pragma unroll
  for (int t = 0; t < 4; ++t)
    #pragma unroll
    for (int u = 0; u < 2; ++u)
      acc[t][u] = (v8f){0.f,0.f,0.f,0.f,0.f,0.f,0.f,0.f};

  stage(0, 0);            // prologue: tile 0 -> buffer 0
  int buf = 0;

  for (int kk = 0; kk < K; kk += BK) {
    const bool hasNext = (kk + BK) < K;
    // Issue next tile's DMA into the other buffer (safe: trailing barrier of
    // the previous iteration guarantees nobody still reads it).
    if (hasNext) stage(kk + BK, buf ^ 1);
    // Prefetch two tiles ahead into near caches (global_prefetch_b8).
    if (kk + 2 * BK < K) {
      __builtin_prefetch(Ab + (long long)(tid >> 1) * lda + kk + 2 * BK, 0, 3);
      __builtin_prefetch(Wb + (long long)(tid >> 1) * ldw + kk + 2 * BK, 0, 3);
    }
#if USE_ASYNC_LDS
    // Drain only the current tile's 4 async loads; next tile stays in flight.
    if (hasNext) __builtin_amdgcn_s_wait_asynccnt(4);
    else         __builtin_amdgcn_s_wait_asynccnt(0);
#endif
    __syncthreads();

    // Fragments (ISA 16-bit A 16x32 / B 32x16 lane striping)
    FragBF afr[4], bfr[2];
    #pragma unroll
    for (int t = 0; t < 4; ++t) {
      int arow = waveM * 64 + t * 16 + l16;
      afr[t].q[0] = *(const uint4*)&Alds[buf][arow][half * 8];
      afr[t].q[1] = *(const uint4*)&Alds[buf][arow][half * 8 + 16];
    }
    #pragma unroll
    for (int u = 0; u < 2; ++u) {
      int nrow = waveN * 32 + u * 16 + l16;
      bfr[u].q[0] = *(const uint4*)&Wlds[buf][nrow][half * 8];
      bfr[u].q[1] = *(const uint4*)&Wlds[buf][nrow][half * 8 + 16];
    }
    #pragma unroll
    for (int t = 0; t < 4; ++t)
      #pragma unroll
      for (int u = 0; u < 2; ++u)
        acc[t][u] = __builtin_amdgcn_wmma_f32_16x16x32_bf16(
            false, afr[t].v, false, bfr[u].v, (short)0, acc[t][u], false, false);

    __syncthreads();      // all waves done reading `buf` before it is reused
    buf ^= 1;
  }

  // Write back: VGPR r -> M = r (lanes 0-15) / 8+r (16-31), N = lane&15
  #pragma unroll
  for (int t = 0; t < 4; ++t)
    #pragma unroll
    for (int u = 0; u < 2; ++u) {
      int baseRow = blockIdx.y * BM + waveM * 64 + t * 16 + half * 8;
      int col     = blockIdx.x * BN + waveN * 32 + u * 16 + l16;
      #pragma unroll
      for (int r = 0; r < 8; ++r) {
        long long off = (long long)(baseRow + r) * ldc + col;
        float val = acc[t][u][r];
        if (accumulate) val += Cb[off];
        Cb[off] = val;
      }
    }
}

// ---------------------------------------------------------------------------
// Block reduction (256 threads = 8 wave32)
// ---------------------------------------------------------------------------
__device__ __forceinline__ float block_sum_256(float v, float* scratch) {
  #pragma unroll
  for (int o = 16; o > 0; o >>= 1) v += __shfl_xor(v, o, 32);
  int lane = threadIdx.x & 31;
  int w    = threadIdx.x >> 5;
  __syncthreads();
  if (lane == 0) scratch[w] = v;
  __syncthreads();
  float total = 0.f;
  #pragma unroll
  for (int i = 0; i < 8; ++i) total += scratch[i];
  return total;
}

__device__ __forceinline__ float sigm(float x) { return 1.f / (1.f + expf(-x)); }

// ---------------------------------------------------------------------------
// Hyper-LSTM cell: LN(v + bih), cell update, LN(c). Also emits bf16 new_h.
// ---------------------------------------------------------------------------
__global__ __launch_bounds__(256)
void hyper_cell_kernel(const float* __restrict__ v, const float* __restrict__ bih,
                       const float* __restrict__ lng, const float* __restrict__ lnb,
                       const float* __restrict__ hyper_c,
                       const float* __restrict__ lncg, const float* __restrict__ lncb,
                       float* __restrict__ out_h, float* __restrict__ out_c,
                       __bf16* __restrict__ out_hb)
{
  __shared__ float red[8];
  const int b = blockIdx.x, j = threadIdx.x;           // Hh = 256
  float gate[4];
  #pragma unroll
  for (int g = 0; g < 4; ++g) {
    float x = v[(long long)b * 1024 + g * 256 + j] + bih[g * 256 + j];
    float S  = block_sum_256(x, red);
    float SS = block_sum_256(x * x, red);
    float mean = S * (1.f / 256.f);
    float var  = (SS - 256.f * mean * mean) * (1.f / 255.f);   // ddof=1
    var = fmaxf(var, 0.f);
    float y = (x - mean) / (sqrtf(var) + 1e-6f);
    gate[g] = y * lng[g * 256 + j] + lnb[g * 256 + j];
  }
  float hi = gate[0], hf = gate[1] + 1.f, hg = gate[2], ho = gate[3];
  float cn = hyper_c[(long long)b * 256 + j] * sigm(hf) + sigm(hi) * tanhf(hg);
  float S  = block_sum_256(cn, red);
  float SS = block_sum_256(cn * cn, red);
  float mean = S * (1.f / 256.f);
  float var  = (SS - 256.f * mean * mean) * (1.f / 255.f);
  var = fmaxf(var, 0.f);
  float cln = (cn - mean) / (sqrtf(var) + 1e-6f) * lncg[j] + lncb[j];
  float hn  = tanhf(cln) * sigm(ho);
  long long o = (long long)b * 256 + j;
  out_c[o]  = cln;
  out_h[o]  = hn;
  out_hb[o] = (__bf16)hn;
}

// p (+ bias) -> bf16, ready to be GEMM A operands
__global__ __launch_bounds__(256)
void bias_pack_kernel(const float* __restrict__ p_h, const float* __restrict__ p_x,
                      const float* __restrict__ p_b,
                      const float* __restrict__ bph, const float* __restrict__ bpx,
                      __bf16* __restrict__ pbh, __bf16* __restrict__ pbx,
                      __bf16* __restrict__ pbb)
{
  int j = threadIdx.x;
  long long o = (long long)blockIdx.x * 256 + j;
  pbh[o] = (__bf16)(p_h[o] + bph[j]);
  pbx[o] = (__bf16)(p_x[o] + bpx[j]);
  pbb[o] = (__bf16)(p_b[o]);
}

// ---------------------------------------------------------------------------
// Main cell: gates = xh*zx + hh*zh + bias + zb (+1 on f), LN per gate over H,
// cell update, LN(c) over H. Block per batch row, 8 cols/thread.
// ---------------------------------------------------------------------------
__global__ __launch_bounds__(256)
void main_cell_kernel(const float* __restrict__ xh, const float* __restrict__ hh,
                      const float* __restrict__ zx, const float* __restrict__ zh,
                      const float* __restrict__ zb, const float* __restrict__ bias,
                      const float* __restrict__ c_in,
                      const float* __restrict__ lng, const float* __restrict__ lnb,
                      const float* __restrict__ lncg, const float* __restrict__ lncb,
                      float* __restrict__ out_h, float* __restrict__ out_c)
{
  __shared__ float red[8];
  const int b = blockIdx.x, t = threadIdx.x;
  float gate[4][8];
  #pragma unroll
  for (int g = 0; g < 4; ++g) {
    float s = 0.f, ss = 0.f;
    #pragma unroll
    for (int i = 0; i < 8; ++i) {
      int col = t + i * 256;
      long long bgx  = (long long)b * 8192 + g * 2048 + col;
      long long zoff = ((long long)g * 512 + b) * 2048 + col;
      float val = xh[bgx] * zx[zoff] + hh[bgx] * zh[zoff]
                + bias[g * 2048 + col] + zb[zoff];
      if (g == 1) val += 1.f;
      gate[g][i] = val;
      s += val; ss += val * val;
    }
    float S  = block_sum_256(s, red);
    float SS = block_sum_256(ss, red);
    float mean = S * (1.f / 2048.f);
    float var  = (SS - 2048.f * mean * mean) * (1.f / 2047.f);  // ddof=1
    var = fmaxf(var, 0.f);
    float inv = 1.f / (sqrtf(var) + 1e-6f);
    #pragma unroll
    for (int i = 0; i < 8; ++i) {
      int col = t + i * 256;
      gate[g][i] = (gate[g][i] - mean) * inv * lng[g * 2048 + col] + lnb[g * 2048 + col];
    }
  }
  float cn[8]; float s = 0.f, ss = 0.f;
  #pragma unroll
  for (int i = 0; i < 8; ++i) {
    int col = t + i * 256;
    float ci = c_in[(long long)b * 2048 + col];
    float val = ci * sigm(gate[1][i]) + sigm(gate[0][i]) * tanhf(gate[2][i]);
    cn[i] = val; s += val; ss += val * val;
  }
  float S  = block_sum_256(s, red);
  float SS = block_sum_256(ss, red);
  float mean = S * (1.f / 2048.f);
  float var  = (SS - 2048.f * mean * mean) * (1.f / 2047.f);
  var = fmaxf(var, 0.f);
  float inv = 1.f / (sqrtf(var) + 1e-6f);
  #pragma unroll
  for (int i = 0; i < 8; ++i) {
    int col = t + i * 256;
    float cln = (cn[i] - mean) * inv * lncg[col] + lncb[col];
    out_c[(long long)b * 2048 + col] = cln;
    out_h[(long long)b * 2048 + col] = tanhf(cln) * sigm(gate[3][i]);
  }
}

// ---------------------------------------------------------------------------
extern "C" void kernel_launch(void* const* d_in, const int* in_sizes, int n_in,
                              void* d_out, int out_size, void* d_ws, size_t ws_size,
                              hipStream_t stream)
{
  const float* x       = (const float*)d_in[0];
  const float* h       = (const float*)d_in[1];
  const float* c       = (const float*)d_in[2];
  const float* hyper_h = (const float*)d_in[3];
  const float* hyper_c = (const float*)d_in[4];
  const float* hc_Wih  = (const float*)d_in[5];
  const float* hc_bih  = (const float*)d_in[6];
  const float* hc_Whh  = (const float*)d_in[7];
  const float* hc_lng  = (const float*)d_in[8];
  const float* hc_lnb  = (const float*)d_in[9];
  const float* hc_lncg = (const float*)d_in[10];
  const float* hc_lncb = (const float*)d_in[11];
  const float* Wph     = (const float*)d_in[12];
  const float* bph     = (const float*)d_in[13];
  const float* Wpx     = (const float*)d_in[14];
  const float* bpx     = (const float*)d_in[15];
  const float* Wpb     = (const float*)d_in[16];
  const float* Wsh     = (const float*)d_in[17];
  const float* Wsx     = (const float*)d_in[18];
  const float* Wsb     = (const float*)d_in[19];
  const float* Wih     = (const float*)d_in[20];
  const float* Whh     = (const float*)d_in[21];
  const float* bias    = (const float*)d_in[22];
  const float* lng     = (const float*)d_in[23];
  const float* lnb     = (const float*)d_in[24];
  const float* lncg    = (const float*)d_in[25];
  const float* lncb    = (const float*)d_in[26];

  const int B = 512, In = 2048, H = 2048, Hh = 256, E = 64;

  float* out_h  = (float*)d_out;
  float* out_c  = out_h  + (long long)B * H;
  float* out_hh = out_c  + (long long)B * H;    // new_hyper_h (f32)
  float* out_hc = out_hh + (long long)B * Hh;   // new_hyper_c

  // ---- workspace: fp32 region first, then bf16 region -------------------
  float* wf  = (float*)d_ws;
  float* v   = wf;  wf += (long long)B * 4 * Hh;   // 512x1024
  float* p_h = wf;  wf += (long long)B * 4 * E;    // 512x256
  float* p_x = wf;  wf += (long long)B * 4 * E;
  float* p_b = wf;  wf += (long long)B * 4 * E;
  float* xh  = wf;  wf += (long long)B * 4 * H;    // 512x8192
  float* hhb = wf;  wf += (long long)B * 4 * H;
  float* zh  = wf;  wf += (long long)4 * B * H;    // 4x512x2048
  float* zx  = wf;  wf += (long long)4 * B * H;
  float* zb  = wf;  wf += (long long)4 * B * H;

  __bf16* bb = (__bf16*)wf;
  __bf16* xb      = bb; bb += (long long)B * In;       // 512x2048
  __bf16* hb      = bb; bb += (long long)B * H;
  __bf16* hyhb    = bb; bb += (long long)B * Hh;       // hyper_h bf16
  __bf16* nhb     = bb; bb += (long long)B * Hh;       // new_hyper_h bf16
  __bf16* pbh     = bb; bb += (long long)B * 4 * E;
  __bf16* pbx     = bb; bb += (long long)B * 4 * E;
  __bf16* pbb     = bb; bb += (long long)B * 4 * E;
  __bf16* hcWihb  = bb; bb += (long long)4 * Hh * (In + H);  // 1024x4096
  __bf16* hcWhhb  = bb; bb += (long long)4 * Hh * Hh;        // 1024x256
  __bf16* Wphb    = bb; bb += (long long)4 * E * Hh;         // 256x256
  __bf16* Wpxb    = bb; bb += (long long)4 * E * Hh;
  __bf16* Wpbb    = bb; bb += (long long)4 * E * Hh;
  __bf16* Wshb    = bb; bb += (long long)4 * H * E;          // 4x2048x64
  __bf16* Wsxb    = bb; bb += (long long)4 * H * E;
  __bf16* Wsbb    = bb; bb += (long long)4 * H * E;
  __bf16* Wihb    = bb; bb += (long long)4 * H * In;         // 8192x2048
  __bf16* Whhb    = bb; bb += (long long)4 * H * H;

  dim3 blk(256);
  auto cvt = [&](const float* s, __bf16* d, long long n) {
    long long n8 = n / 8;
    cvt_f32_bf16_kernel<<<dim3((unsigned)(n8 / 256)), blk, 0, stream>>>(s, d, n8);
  };

  // ---- one-shot fp32 -> bf16 conversions --------------------------------
  cvt(x,       xb,     (long long)B * In);
  cvt(h,       hb,     (long long)B * H);
  cvt(hyper_h, hyhb,   (long long)B * Hh);
  cvt(hc_Wih,  hcWihb, (long long)4 * Hh * (In + H));
  cvt(hc_Whh,  hcWhhb, (long long)4 * Hh * Hh);
  cvt(Wph,     Wphb,   (long long)4 * E * Hh);
  cvt(Wpx,     Wpxb,   (long long)4 * E * Hh);
  cvt(Wpb,     Wpbb,   (long long)4 * E * Hh);
  cvt(Wsh,     Wshb,   (long long)4 * H * E);
  cvt(Wsx,     Wsxb,   (long long)4 * H * E);
  cvt(Wsb,     Wsbb,   (long long)4 * H * E);
  cvt(Wih,     Wihb,   (long long)4 * H * In);
  cvt(Whh,     Whhb,   (long long)4 * H * H);

  dim3 gV(4 * Hh / BN, B / BM, 1);      // (8, 4)
  dim3 gP(4 * E  / BN, B / BM, 1);      // (2, 4)
  dim3 gZ(H      / BN, B / BM, 4);      // (16, 4, 4)
  dim3 gG(4 * H  / BN, B / BM, 1);      // (64, 4)

  // v = x @ hc_Wih[:, :In]^T + h @ hc_Wih[:, In:]^T + hyper_h @ hc_Whh^T
  gemm_bf16_wmma<<<gV, blk, 0, stream>>>(xb, In, 0, hcWihb, In + H, 0,
                                         v, 4 * Hh, 0, In, 0);
  gemm_bf16_wmma<<<gV, blk, 0, stream>>>(hb, H, 0, hcWihb + In, In + H, 0,
                                         v, 4 * Hh, 0, H, 1);
  gemm_bf16_wmma<<<gV, blk, 0, stream>>>(hyhb, Hh, 0, hcWhhb, Hh, 0,
                                         v, 4 * Hh, 0, Hh, 1);

  hyper_cell_kernel<<<B, blk, 0, stream>>>(v, hc_bih, hc_lng, hc_lnb,
                                           hyper_c, hc_lncg, hc_lncb,
                                           out_hh, out_hc, nhb);

  // p_* = new_hyper_h @ Wp*^T   (Wp flat is (256 x 256) row-major)
  gemm_bf16_wmma<<<gP, blk, 0, stream>>>(nhb, Hh, 0, Wphb, Hh, 0,
                                         p_h, 4 * E, 0, Hh, 0);
  gemm_bf16_wmma<<<gP, blk, 0, stream>>>(nhb, Hh, 0, Wpxb, Hh, 0,
                                         p_x, 4 * E, 0, Hh, 0);
  gemm_bf16_wmma<<<gP, blk, 0, stream>>>(nhb, Hh, 0, Wpbb, Hh, 0,
                                         p_b, 4 * E, 0, Hh, 0);

  bias_pack_kernel<<<B, blk, 0, stream>>>(p_h, p_x, p_b, bph, bpx, pbh, pbx, pbb);

  // z*[g] = p*[:, g*64:(g+1)*64] @ Ws*[g]^T   (batched over g via grid.z)
  gemm_bf16_wmma<<<gZ, blk, 0, stream>>>(pbh, 4 * E, E, Wshb, E, (long long)H * E,
                                         zh, H, (long long)B * H, E, 0);
  gemm_bf16_wmma<<<gZ, blk, 0, stream>>>(pbx, 4 * E, E, Wsxb, E, (long long)H * E,
                                         zx, H, (long long)B * H, E, 0);
  gemm_bf16_wmma<<<gZ, blk, 0, stream>>>(pbb, 4 * E, E, Wsbb, E, (long long)H * E,
                                         zb, H, (long long)B * H, E, 0);

  // xh = x @ Wih^T,  hh = h @ Whh^T   (the two 17-GFLOP GEMMs)
  gemm_bf16_wmma<<<gG, blk, 0, stream>>>(xb, In, 0, Wihb, In, 0,
                                         xh, 4 * H, 0, In, 0);
  gemm_bf16_wmma<<<gG, blk, 0, stream>>>(hb, H, 0, Whhb, H, 0,
                                         hhb, 4 * H, 0, H, 0);

  main_cell_kernel<<<B, blk, 0, stream>>>(xh, hhb, zx, zh, zb, bias, c,
                                          lng, lnb, lncg, lncb, out_h, out_c);
}
